// EquivariantBlock_61211873903540
// MI455X (gfx1250) — matrix operand order repaired and checked by reading, not compiled, not run on hardware
//
#include <hip/hip_runtime.h>

// ---------------- types ----------------
typedef __attribute__((ext_vector_type(16))) _Float16 v16h;
typedef __attribute__((ext_vector_type(8)))  float    v8f;

__device__ __forceinline__ float fast_rcp(float x) { return __builtin_amdgcn_rcpf(x); }
__device__ __forceinline__ float silu_f(float x) { return x * fast_rcp(1.0f + __expf(-x)); }

__device__ __forceinline__ unsigned enc_f(float f) {
  unsigned u = __float_as_uint(f);
  return (u & 0x80000000u) ? ~u : (u | 0x80000000u);
}
__device__ __forceinline__ float dec_f(unsigned u) {
  unsigned v = (u & 0x80000000u) ? (u & 0x7FFFFFFFu) : ~u;
  return __uint_as_float(v);
}

__device__ __forceinline__ float wave_sum(float s) {
  #pragma unroll
  for (int o = 16; o > 0; o >>= 1) s += __shfl_xor(s, o);
  return s;
}

// ---------------- A-row gather functors ----------------
struct ADirectH {  // f16 source: no cvt in inner loop
  const _Float16* p; int ld;
  struct Row { const _Float16* p;
    __device__ _Float16 operator()(int k) const { return p[k]; } };
  __device__ Row row(int m) const { return Row{ p + (size_t)m * ld }; }
};
struct AConcat2 {  // fp32 concat (dist | edge_attr)
  const float* a; const float* b; int lda; int ldb; int split;
  struct Row { const float* a; const float* b; int split;
    __device__ float operator()(int k) const { return k < split ? a[k] : b[k - split]; } };
  __device__ Row row(int m) const {
    return Row{ a + (size_t)m * lda, b + (size_t)m * ldb, split };
  }
};
struct ASum2 {     // h_node[row] + h_node[col]
  const float* p; int ld; const int* i0; const int* i1;
  struct Row { const float* a; const float* b;
    __device__ float operator()(int k) const { return a[k] + b[k]; } };
  __device__ Row row(int m) const {
    return Row{ p + (size_t)i0[m] * ld, p + (size_t)i1[m] * ld };
  }
};
struct AEq { // concat(h_out[row](256), h_out[col](256), h_edge(64), dist(64)) -> K=640
  const float* hout; const float* hedge; const float* dist; const int* r; const int* c;
  struct Row { const float* a; const float* b; const float* hb; const float* db;
    __device__ float operator()(int k) const {
      if (k < 256) return a[k];
      if (k < 512) return b[k - 256];
      if (k < 576) return hb[k - 512];
      return db[k - 576];
    } };
  __device__ Row row(int m) const {
    return Row{ hout + (size_t)r[m] * 256, hout + (size_t)c[m] * 256,
                hedge + (size_t)m * 64, dist + (size_t)m * 64 };
  }
};

// ---------------- weight swizzle: f32 [K,N] -> f16 B-fragment layout ----------------
// Wsw[((k>>5)*N + n)*32 + (k&31)] = (f16) W[k*N + n]
// A lane's 16-half B fragment becomes 32 contiguous bytes -> two b128 loads.
__global__ __launch_bounds__(256) void k_wswz(
    const float* __restrict__ W, _Float16* __restrict__ out, int N, long tot)
{
  long i = (long)blockIdx.x * 256 + threadIdx.x;
  if (i >= tot) return;
  int k = (int)(i / N), n = (int)(i % N);
  out[(((size_t)(k >> 5) * N + n) << 5) + (k & 31)] = (_Float16)W[i];
}

// ---------------- generic WMMA gather-GEMM ----------------
// Out[M,NDIM] = act( A[M,KDIM] @ W[KDIM,NDIM] + bias )
// A via gather functor (fp32 or f16); W pre-swizzled f16; f32 accumulate.
// One wave per (mSuper,nTile); MT m-tiles per wave -> B fragment reused MT times.
template<int KDIM, int NDIM, int MT, bool OSILU, typename TOUT, typename AF>
__global__ __launch_bounds__(256) void gemm_wmma(
    AF af, const _Float16* __restrict__ W, const float* __restrict__ bias,
    TOUT* __restrict__ out, int M)
{
  const int numNT = NDIM / 16;
  const int wid  = (int)(threadIdx.x >> 5);
  const int lane = (int)(threadIdx.x & 31);
  const int half = lane >> 4;
  const int lm   = lane & 15;
  long gw = (long)blockIdx.x * 8 + wid;
  int mSuper = (int)(gw / numNT);
  int nTile  = (int)(gw % numNT);
  int mBase  = mSuper * (16 * MT);
  if (mBase >= M) return;
  int ncol = nTile * 16 + lm;
  const bool full = (mBase + 16 * MT) <= M;

  typename AF::Row rows[MT];
  #pragma unroll
  for (int mt = 0; mt < MT; mt++) {
    int r = mBase + mt * 16 + lm;
    if (r >= M) r = M - 1;
    rows[mt] = af.row(r);
  }

  v8f acc[MT];
  #pragma unroll
  for (int mt = 0; mt < MT; mt++)
    #pragma unroll
    for (int r = 0; r < 8; r++) acc[mt][r] = 0.0f;

  for (int k0 = 0; k0 < KDIM; k0 += 32) {
    // B fragment: 16 contiguous halves per lane from swizzled weights
    const _Float16* wp = W + (((size_t)(k0 >> 5) * NDIM + ncol) << 5) + half * 16;
    v16h b;
    #pragma unroll
    for (int t = 0; t < 16; t++) b[t] = wp[t];
    #pragma unroll
    for (int mt = 0; mt < MT; mt++) {
      // A fragment (16x32): lanes 0-15 row=lm K:[k0..k0+15];
      //                     lanes 16-31 row=lm K:[k0+16..k0+31]
      v16h a;
      #pragma unroll
      for (int t = 0; t < 8; t++)
        a[t] = (_Float16)rows[mt](k0 + half * 16 + t);
      #pragma unroll
      for (int t = 0; t < 8; t++)
        a[8 + t] = (_Float16)rows[mt](k0 + half * 16 + 8 + t);
      acc[mt] = __builtin_amdgcn_wmma_f32_16x16x32_f16(
          false, a, false, b, (short)0, acc[mt], false, false);
    }
  }

  float bv = bias ? bias[ncol] : 0.0f;
  if (full) {
    #pragma unroll
    for (int mt = 0; mt < MT; mt++) {
      #pragma unroll
      for (int r = 0; r < 8; r++) {
        int row = mBase + mt * 16 + half * 8 + r;
        float v = acc[mt][r] + bv;
        if (OSILU) v = silu_f(v);
        out[(size_t)row * NDIM + ncol] = (TOUT)v;
      }
    }
  } else {
    #pragma unroll
    for (int mt = 0; mt < MT; mt++) {
      #pragma unroll
      for (int r = 0; r < 8; r++) {
        int row = mBase + mt * 16 + half * 8 + r;
        if (row < M) {
          float v = acc[mt][r] + bv;
          if (OSILU) v = silu_f(v);
          out[(size_t)row * NDIM + ncol] = (TOUT)v;
        }
      }
    }
  }
}

// ---------------- elementwise / wave-per-row kernels ----------------
__global__ __launch_bounds__(256) void k_zero(float* __restrict__ p, long n) {
  long i = (long)blockIdx.x * 256 + threadIdx.x;
  if (i < n) p[i] = 0.0f;
}

__global__ __launch_bounds__(256) void k_pos_init(const float* __restrict__ pos,
                                                  float* __restrict__ out, long n) {
  long i = (long)blockIdx.x * 256 + threadIdx.x;
  if (i < n) out[i] = pos[i];
}

// y = (f16) silu(x)
__global__ __launch_bounds__(256) void k_silu_h(const float* __restrict__ x,
                                                _Float16* __restrict__ y, long n) {
  long i = (long)blockIdx.x * 256 + threadIdx.x;
  if (i < n) y[i] = (_Float16)silu_f(x[i]);
}

// GBF: d2, time-conditioned affine, Gaussian basis -> dist[E,64]
__global__ __launch_bounds__(256) void k_prep_edge(
    const float* __restrict__ pos, const float* __restrict__ ete,
    const int* __restrict__ ei,
    const float* __restrict__ mulw, const float* __restrict__ mulb,
    const float* __restrict__ biasw, const float* __restrict__ biasb,
    const float* __restrict__ means, const float* __restrict__ stds,
    float* __restrict__ dist, int E)
{
  int wid = threadIdx.x >> 5, lane = threadIdx.x & 31;
  int e = blockIdx.x * 8 + wid;
  if (e >= E) return;
  int r = ei[e], c = ei[E + e];
  float dx = pos[(size_t)r*3+0] - pos[(size_t)c*3+0];
  float dy = pos[(size_t)r*3+1] - pos[(size_t)c*3+1];
  float dz = pos[(size_t)r*3+2] - pos[(size_t)c*3+2];
  float d2 = dx*dx + dy*dy + dz*dz;
  const float* t = ete + (size_t)e * 256;
  float sm = 0.0f, sb = 0.0f;
  #pragma unroll
  for (int j = 0; j < 8; j++) {
    float tv = t[lane + 32*j];
    sm += tv * mulw[lane + 32*j];
    sb += tv * biasw[lane + 32*j];
  }
  sm = wave_sum(sm); sb = wave_sum(sb);
  float x = (sm + mulb[0]) * d2 + (sb + biasb[0]);
  #pragma unroll
  for (int j = 0; j < 2; j++) {
    int cc = lane + 32*j;
    float sd = fabsf(stds[cc]) + 1e-5f;
    float rs = fast_rcp(sd);
    float pre = (x - means[cc]) * rs;
    dist[(size_t)e*64 + cc] = __expf(-0.5f * pre * pre) * 0.3989422804014327f * rs;
  }
}

// h_mod = mod(ln(h), nt_sh_msa, nt_sc_msa)  -> f16  (wave per node, 256 wide)
__global__ __launch_bounds__(256) void k_node_mod(
    const float* __restrict__ h, const float* __restrict__ nt,
    _Float16* __restrict__ out, int Nn)
{
  int wid = threadIdx.x >> 5, lane = threadIdx.x & 31;
  int n = blockIdx.x * 8 + wid;
  if (n >= Nn) return;
  const float* hr = h + (size_t)n * 256;
  float vals[8], s = 0.0f, s2 = 0.0f;
  #pragma unroll
  for (int j = 0; j < 8; j++) { float v = hr[lane + 32*j]; vals[j] = v; s += v; s2 += v*v; }
  s = wave_sum(s); s2 = wave_sum(s2);
  float mean = s * (1.0f/256.0f);
  float inv = rsqrtf(s2 * (1.0f/256.0f) - mean*mean + 1e-6f);
  const float* ntr = nt + (size_t)n * 1536;
  #pragma unroll
  for (int j = 0; j < 8; j++) {
    int c = lane + 32*j;
    out[(size_t)n*256 + c] =
        (_Float16)((vals[j]-mean)*inv * (1.0f + ntr[256 + c]) + ntr[c]);
  }
}

// edge emb -> mod(ln(.), et_sh_msa, et_sc_msa) -> f16  (wave per edge, 64 wide)
__global__ __launch_bounds__(256) void k_edge_mod(
    const float* __restrict__ ea, const float* __restrict__ et,
    _Float16* __restrict__ out, int E)
{
  int wid = threadIdx.x >> 5, lane = threadIdx.x & 31;
  int e = blockIdx.x * 8 + wid;
  if (e >= E) return;
  const float* er = ea + (size_t)e * 64;
  float v0 = er[lane], v1 = er[lane + 32];
  float s = wave_sum(v0 + v1);
  float s2 = wave_sum(v0*v0 + v1*v1);
  float mean = s * (1.0f/64.0f);
  float inv = rsqrtf(s2 * (1.0f/64.0f) - mean*mean + 1e-6f);
  const float* etr = et + (size_t)e * 384;
  out[(size_t)e*64 + lane] =
      (_Float16)((v0-mean)*inv * (1.0f + etr[64 + lane]) + etr[lane]);
  out[(size_t)e*64 + lane + 32] =
      (_Float16)((v1-mean)*inv * (1.0f + etr[64 + lane + 32]) + etr[lane + 32]);
}

// alpha[e,h] = dot(q[col],k[row],e0)/sqrt(32); segment-max via u32 atomicMax
__global__ __launch_bounds__(256) void k_alpha(
    const float* __restrict__ q, const float* __restrict__ kk,
    const float* __restrict__ e0, const int* __restrict__ ei,
    float* __restrict__ alpha, unsigned* __restrict__ amax, int E)
{
  int wid = threadIdx.x >> 5, lane = threadIdx.x & 31;
  int e = blockIdx.x * 8 + wid;
  if (e >= E) return;
  int r = ei[e], c = ei[E + e];
  const float* qr = q + (size_t)c * 256;
  const float* kr = kk + (size_t)r * 256;
  const float* er = e0 + (size_t)e * 256;
  #pragma unroll
  for (int h = 0; h < 8; h++) {
    int o = h*32 + lane;
    float s = wave_sum(qr[o] * kr[o] * er[o]) * 0.17677669529663687f;
    if (lane == 0) {
      alpha[(size_t)e*8 + h] = s;
      atomicMax(&amax[(size_t)c*8 + h], enc_f(s));
    }
  }
}

// ex = exp(alpha - amax[col]); denom += ex   (thread per (e,h))
__global__ __launch_bounds__(256) void k_exp(
    float* __restrict__ alpha, const unsigned* __restrict__ amax,
    float* __restrict__ den, const int* __restrict__ ei, int E)
{
  long i = (long)blockIdx.x * 256 + threadIdx.x;
  if (i >= (long)E * 8) return;
  int e = (int)(i >> 3), h = (int)(i & 7);
  int c = ei[E + e];
  float ex = __expf(alpha[i] - dec_f(amax[(size_t)c*8 + h]));
  alpha[i] = ex;
  atomicAdd(&den[(size_t)c*8 + h], ex);
}

// msg = v[row]*e1*attn ; scatter-add into h_node[col]   (wave per edge)
__global__ __launch_bounds__(256) void k_msg(
    const float* __restrict__ v, const float* __restrict__ e1,
    const float* __restrict__ ex, const float* __restrict__ den,
    const int* __restrict__ ei, float* __restrict__ hnode, int E)
{
  int wid = threadIdx.x >> 5, lane = threadIdx.x & 31;
  int e = blockIdx.x * 8 + wid;
  if (e >= E) return;
  int r = ei[e], c = ei[E + e];
  const float* vr = v + (size_t)r * 256;
  const float* er = e1 + (size_t)e * 256;
  #pragma unroll
  for (int h = 0; h < 8; h++) {
    float attn = ex[(size_t)e*8 + h] * fast_rcp(den[(size_t)c*8 + h] + 1e-16f);
    int o = h*32 + lane;
    atomicAdd(&hnode[(size_t)c*256 + o], vr[o] * er[o] * attn);
  }
}

// node residual + LN + mlp-mod + mask -> f16  (wave per node)
__global__ __launch_bounds__(256) void k_node_res(
    const float* __restrict__ hin, const float* __restrict__ nt,
    const float* __restrict__ hnode, const float* __restrict__ mask,
    _Float16* __restrict__ out, int Nn)
{
  int wid = threadIdx.x >> 5, lane = threadIdx.x & 31;
  int n = blockIdx.x * 8 + wid;
  if (n >= Nn) return;
  const float* ntr = nt + (size_t)n * 1536;
  float vals[8], s = 0.0f, s2 = 0.0f;
  #pragma unroll
  for (int j = 0; j < 8; j++) {
    int c = lane + 32*j;
    float v = hin[(size_t)n*256 + c] + ntr[512 + c] * hnode[(size_t)n*256 + c];
    vals[j] = v; s += v; s2 += v*v;
  }
  s = wave_sum(s); s2 = wave_sum(s2);
  float mean = s * (1.0f/256.0f);
  float inv = rsqrtf(s2 * (1.0f/256.0f) - mean*mean + 1e-6f);
  float mk = mask[n];
  #pragma unroll
  for (int j = 0; j < 8; j++) {
    int c = lane + 32*j;
    out[(size_t)n*256 + c] =
        (_Float16)(((vals[j]-mean)*inv * (1.0f + ntr[1024 + c]) + ntr[768 + c]) * mk);
  }
}

__global__ __launch_bounds__(256) void k_hout(
    const _Float16* __restrict__ hmod, const float* __restrict__ nt,
    const float* __restrict__ ffn, const float* __restrict__ mask,
    float* __restrict__ out, long total)
{
  long i = (long)blockIdx.x * 256 + threadIdx.x;
  if (i >= total) return;
  long n = i >> 8; int c = (int)(i & 255);
  out[i] = ((float)hmod[i] + nt[n*1536 + 1280 + c] * ffn[i]) * mask[n];
}

// edge residual + LN + mlp-mod -> f16  (wave per edge, 64 wide)
__global__ __launch_bounds__(256) void k_edge_res(
    const float* __restrict__ eain, const float* __restrict__ et,
    const float* __restrict__ n2e, _Float16* __restrict__ out, int E)
{
  int wid = threadIdx.x >> 5, lane = threadIdx.x & 31;
  int e = blockIdx.x * 8 + wid;
  if (e >= E) return;
  const float* etr = et + (size_t)e * 384;
  float v0 = eain[(size_t)e*64 + lane]      + etr[128 + lane]      * n2e[(size_t)e*64 + lane];
  float v1 = eain[(size_t)e*64 + lane + 32] + etr[128 + lane + 32] * n2e[(size_t)e*64 + lane + 32];
  float s = wave_sum(v0 + v1);
  float s2 = wave_sum(v0*v0 + v1*v1);
  float mean = s * (1.0f/64.0f);
  float inv = rsqrtf(s2 * (1.0f/64.0f) - mean*mean + 1e-6f);
  out[(size_t)e*64 + lane] =
      (_Float16)((v0-mean)*inv * (1.0f + etr[256 + lane]) + etr[192 + lane]);
  out[(size_t)e*64 + lane + 32] =
      (_Float16)((v1-mean)*inv * (1.0f + etr[256 + lane + 32]) + etr[192 + lane + 32]);
}

__global__ __launch_bounds__(256) void k_eout(
    const _Float16* __restrict__ hemod, const float* __restrict__ et,
    const float* __restrict__ ffe, float* __restrict__ out, long total)
{
  long i = (long)blockIdx.x * 256 + threadIdx.x;
  if (i >= total) return;
  long e = i >> 6; int c = (int)(i & 63);
  out[i] = (float)hemod[i] + et[e*384 + 320 + c] * ffe[i];
}

// eq invariant: LN + time-mod (shift/scale from f16 ts) -> f16  (wave per edge, 256 wide)
__global__ __launch_bounds__(256) void k_eq_mod(
    const float* __restrict__ x, const _Float16* __restrict__ ts,
    _Float16* __restrict__ out, int E)
{
  int wid = threadIdx.x >> 5, lane = threadIdx.x & 31;
  int e = blockIdx.x * 8 + wid;
  if (e >= E) return;
  const float* xr = x + (size_t)e * 256;
  float vals[8], s = 0.0f, s2 = 0.0f;
  #pragma unroll
  for (int j = 0; j < 8; j++) { float v = xr[lane + 32*j]; vals[j] = v; s += v; s2 += v*v; }
  s = wave_sum(s); s2 = wave_sum(s2);
  float mean = s * (1.0f/256.0f);
  float inv = rsqrtf(s2 * (1.0f/256.0f) - mean*mean + 1e-6f);
  const _Float16* tr = ts + (size_t)e * 512;
  #pragma unroll
  for (int j = 0; j < 8; j++) {
    int c = lane + 32*j;
    out[(size_t)e*256 + c] =
        (_Float16)((vals[j]-mean)*inv * (1.0f + (float)tr[256 + c]) + (float)tr[c]);
  }
}

// eq_c2 dot + tanh + CoorsNorm + scatter into pos_out[row]  (wave per edge)
__global__ __launch_bounds__(256) void k_eq_final(
    const _Float16* __restrict__ inv2, const float* __restrict__ c2w,
    const float* __restrict__ pos, const int* __restrict__ ei,
    const float* __restrict__ scale, float* __restrict__ pos_out, int E)
{
  int wid = threadIdx.x >> 5, lane = threadIdx.x & 31;
  int e = blockIdx.x * 8 + wid;
  if (e >= E) return;
  const _Float16* xr = inv2 + (size_t)e * 256;
  float s = 0.0f;
  #pragma unroll
  for (int j = 0; j < 8; j++) s += (float)xr[lane + 32*j] * c2w[lane + 32*j];
  s = wave_sum(s);
  float invv = tanhf(s);
  int r = ei[e], c = ei[E + e];
  float dx = pos[(size_t)r*3+0] - pos[(size_t)c*3+0];
  float dy = pos[(size_t)r*3+1] - pos[(size_t)c*3+1];
  float dz = pos[(size_t)r*3+2] - pos[(size_t)c*3+2];
  float nrm = sqrtf(dx*dx + dy*dy + dz*dz);
  float f = scale[0] * fast_rcp(fmaxf(nrm, 1e-8f)) * invv;
  if (lane == 0) atomicAdd(&pos_out[(size_t)r*3+0], dx * f);
  if (lane == 1) atomicAdd(&pos_out[(size_t)r*3+1], dy * f);
  if (lane == 2) atomicAdd(&pos_out[(size_t)r*3+2], dz * f);
}

// ---------------- host ----------------
static inline unsigned gemm_blocks(int M, int NDIM, int MT) {
  long mt = ((long)M + 16L*MT - 1) / (16L*MT);
  long waves = mt * (NDIM / 16);
  return (unsigned)((waves + 7) / 8);
}

extern "C" void kernel_launch(void* const* d_in, const int* in_sizes, int n_in,
                              void* d_out, int out_size, void* d_ws, size_t ws_size,
                              hipStream_t stream)
{
  const float* pos  = (const float*)d_in[0];
  const float* hin  = (const float*)d_in[1];
  const float* eain = (const float*)d_in[2];
  const int*   ei   = (const int*)d_in[3];
  const float* mask = (const float*)d_in[4];
  const float* nte  = (const float*)d_in[5];
  const float* ete  = (const float*)d_in[6];
  const float* P[38];
  for (int i = 0; i < 38; i++) P[i] = (const float*)d_in[7 + i];
  enum { GBF_MULW=0, GBF_MULB, GBF_BW, GBF_BB, GBF_MEANS, GBF_STDS,
         EEMB_W, EEMB_B, N2E_W, N2E_B, Q_W, Q_B, K_W, K_B, V_W, V_B,
         E0_W, E1_W, FF1_W, FF1_B, FF2_W, FF2_B, FF3_W, FF3_B, FF4_W, FF4_B,
         NT_W, NT_B, ET_W, ET_B, EQ_SCALE, EQT_W, EQT_B, EQI_W, EQI_B,
         EQC1_W, EQC1_B, EQC2_W };

  const int Nn = in_sizes[0] / 3;
  const int E  = in_sizes[3] / 2;
  float* ws = (float*)d_ws;

  // workspace layout (units of float; half buffers use elems/2 floats)
  size_t off = 0;
  auto alloc = [&](size_t n) { size_t o = off; off += n; return o; };
  size_t B_dist  = alloc((size_t)E * 64);     // f32 [E,64]
  size_t B_et    = alloc((size_t)E * 384);    // f32 [E,384]
  size_t B_nt    = alloc((size_t)Nn * 1536);  // f32 [N,1536]
  size_t B_q     = alloc((size_t)Nn * 256);
  size_t B_k     = alloc((size_t)Nn * 256);
  size_t B_v     = alloc((size_t)Nn * 256);
  size_t B_eemb  = alloc((size_t)E * 64);     // f32 edge emb (pre-LN)
  size_t B_e0    = alloc((size_t)E * 256);    // f32; reused later as eq_in out
  size_t B_e1    = alloc((size_t)E * 256);    // f32; reused later: eqmh + inv2h (f16)
  size_t B_ex    = alloc((size_t)E * 8);
  size_t B_amax  = alloc((size_t)Nn * 8);     // amax|den|hnode contiguous -> one zero pass
  size_t B_den   = alloc((size_t)Nn * 8);
  size_t B_hnode = alloc((size_t)Nn * 256);
  size_t B_n2e   = alloc((size_t)E * 64);
  size_t B_ffn   = alloc((size_t)Nn * 256);
  size_t B_ffe   = alloc((size_t)E * 64);
  // f16 activation buffers
  size_t B_snte  = alloc((size_t)Nn * 128);   // silu(nte)  f16 [N,256]
  size_t B_sete  = alloc((size_t)E * 128);    // silu(ete)  f16 [E,256]
  size_t B_hmodh = alloc((size_t)Nn * 128);   // f16 [N,256]
  size_t B_emodh = alloc((size_t)E * 32);     // f16 [E,64]
  size_t B_hnmodh= alloc((size_t)Nn * 128);   // f16 [N,256]
  size_t B_ff1h  = alloc((size_t)Nn * 256);   // f16 [N,512]
  size_t B_ff3h  = alloc((size_t)E * 64);     // f16 [E,128]
  size_t B_tsh   = alloc((size_t)E * 256);    // f16 [E,512]
  size_t B_hemodh= alloc((size_t)E * 32);     // f16 [E,64]
  size_t B_wh    = alloc(720000);             // f16 swizzled weights (~1.39M halves)
  (void)ws_size;

  _Float16* snte   = (_Float16*)(ws + B_snte);
  _Float16* sete   = (_Float16*)(ws + B_sete);
  _Float16* hmodh  = (_Float16*)(ws + B_hmodh);
  _Float16* emodh  = (_Float16*)(ws + B_emodh);
  _Float16* hnmodh = (_Float16*)(ws + B_hnmodh);
  _Float16* ff1h   = (_Float16*)(ws + B_ff1h);
  _Float16* ff3h   = (_Float16*)(ws + B_ff3h);
  _Float16* tsh    = (_Float16*)(ws + B_tsh);
  _Float16* hemodh = (_Float16*)(ws + B_hemodh);
  _Float16* eqmh   = (_Float16*)(ws + B_e1);                       // reuse of B_e1
  _Float16* inv2h  = (_Float16*)(ws + B_e1 + (size_t)E * 128);     // (dead after k_msg)

  float* out_h    = (float*)d_out;                 // [N,256]
  float* out_edge = out_h + (size_t)Nn * 256;      // [E,64]
  float* out_pos  = out_edge + (size_t)E * 64;     // [N,3]

  const int eb = (E + 7) / 8;
  const int nb = (Nn + 7) / 8;

  // 0) zero accumulators
  {
    long tot = (long)Nn * (8 + 8 + 256);
    k_zero<<<(unsigned)((tot + 255) / 256), 256, 0, stream>>>(ws + B_amax, tot);
  }

  // 1) pre-swizzle all GEMM weights to f16 B-fragment layout (L2-resident, ~2.8 MB)
  _Float16* wh = (_Float16*)(ws + B_wh);
  size_t woff = 0;
  auto swz = [&](const float* W, int K, int N) -> _Float16* {
    _Float16* dst = wh + woff;
    long tot = (long)K * N;
    woff += (size_t)tot;
    k_wswz<<<(unsigned)((tot + 255) / 256), 256, 0, stream>>>(W, dst, N, tot);
    return dst;
  };
  _Float16* W_nt   = swz(P[NT_W],   256, 1536);
  _Float16* W_et   = swz(P[ET_W],   256, 384);
  _Float16* W_eqt  = swz(P[EQT_W],  256, 512);
  _Float16* W_eemb = swz(P[EEMB_W], 128, 64);
  _Float16* W_q    = swz(P[Q_W],    256, 256);
  _Float16* W_k    = swz(P[K_W],    256, 256);
  _Float16* W_v    = swz(P[V_W],    256, 256);
  _Float16* W_e0   = swz(P[E0_W],   64, 256);
  _Float16* W_e1   = swz(P[E1_W],   64, 256);
  _Float16* W_n2e  = swz(P[N2E_W],  256, 64);
  _Float16* W_ff1  = swz(P[FF1_W],  256, 512);
  _Float16* W_ff2  = swz(P[FF2_W],  512, 256);
  _Float16* W_ff3  = swz(P[FF3_W],  64, 128);
  _Float16* W_ff4  = swz(P[FF4_W],  128, 64);
  _Float16* W_eqi  = swz(P[EQI_W],  640, 256);
  _Float16* W_eqc1 = swz(P[EQC1_W], 256, 256);

  // 2) silu(time_emb) once -> f16
  k_silu_h<<<(unsigned)(((long)Nn*256 + 255)/256), 256, 0, stream>>>(nte, snte, (long)Nn*256);
  k_silu_h<<<(unsigned)(((long)E*256 + 255)/256), 256, 0, stream>>>(ete, sete, (long)E*256);

  // 3) GBF distance features
  k_prep_edge<<<eb, 256, 0, stream>>>(pos, ete, ei, P[GBF_MULW], P[GBF_MULB],
      P[GBF_BW], P[GBF_BB], P[GBF_MEANS], P[GBF_STDS], ws + B_dist, E);

  // 4) time modulations
  gemm_wmma<256,1536,4,false,float><<<gemm_blocks(Nn,1536,4),256,0,stream>>>(
      ADirectH{snte, 256}, W_nt, P[NT_B], ws + B_nt, Nn);
  gemm_wmma<256,384,4,false,float><<<gemm_blocks(E,384,4),256,0,stream>>>(
      ADirectH{sete, 256}, W_et, P[ET_B], ws + B_et, E);
  gemm_wmma<256,512,4,false,_Float16><<<gemm_blocks(E,512,4),256,0,stream>>>(
      ADirectH{sete, 256}, W_eqt, P[EQT_B], tsh, E);

  // 5) node LN+mod; edge embedding; edge LN+mod
  k_node_mod<<<nb, 256, 0, stream>>>(hin, ws + B_nt, hmodh, Nn);
  gemm_wmma<128,64,4,false,float><<<gemm_blocks(E,64,4),256,0,stream>>>(
      AConcat2{ws + B_dist, eain, 64, 64, 64}, W_eemb, P[EEMB_B], ws + B_eemb, E);
  k_edge_mod<<<eb, 256, 0, stream>>>(ws + B_eemb, ws + B_et, emodh, E);

  // 6) q/k/v, e0/e1 projections
  gemm_wmma<256,256,4,false,float><<<gemm_blocks(Nn,256,4),256,0,stream>>>(
      ADirectH{hmodh, 256}, W_q, P[Q_B], ws + B_q, Nn);
  gemm_wmma<256,256,4,false,float><<<gemm_blocks(Nn,256,4),256,0,stream>>>(
      ADirectH{hmodh, 256}, W_k, P[K_B], ws + B_k, Nn);
  gemm_wmma<256,256,4,false,float><<<gemm_blocks(Nn,256,4),256,0,stream>>>(
      ADirectH{hmodh, 256}, W_v, P[V_B], ws + B_v, Nn);
  gemm_wmma<64,256,4,false,float><<<gemm_blocks(E,256,4),256,0,stream>>>(
      ADirectH{emodh, 64}, W_e0, nullptr, ws + B_e0, E);
  gemm_wmma<64,256,4,false,float><<<gemm_blocks(E,256,4),256,0,stream>>>(
      ADirectH{emodh, 64}, W_e1, nullptr, ws + B_e1, E);

  // 7) edge softmax attention + aggregation
  k_alpha<<<eb, 256, 0, stream>>>(ws + B_q, ws + B_k, ws + B_e0, ei,
      ws + B_ex, (unsigned*)(ws + B_amax), E);
  k_exp<<<(unsigned)(((long)E*8 + 255)/256), 256, 0, stream>>>(
      ws + B_ex, (const unsigned*)(ws + B_amax), ws + B_den, ei, E);
  k_msg<<<eb, 256, 0, stream>>>(ws + B_v, ws + B_e1, ws + B_ex, ws + B_den,
      ei, ws + B_hnode, E);

  // 8) node->edge message
  gemm_wmma<256,64,4,false,float><<<gemm_blocks(E,64,4),256,0,stream>>>(
      ASum2{ws + B_hnode, 256, ei, ei + E}, W_n2e, P[N2E_B], ws + B_n2e, E);

  // 9) node residual + FF
  k_node_res<<<nb, 256, 0, stream>>>(hin, ws + B_nt, ws + B_hnode, mask, hnmodh, Nn);
  gemm_wmma<256,512,4,true,_Float16><<<gemm_blocks(Nn,512,4),256,0,stream>>>(
      ADirectH{hnmodh, 256}, W_ff1, P[FF1_B], ff1h, Nn);
  gemm_wmma<512,256,4,false,float><<<gemm_blocks(Nn,256,4),256,0,stream>>>(
      ADirectH{ff1h, 512}, W_ff2, P[FF2_B], ws + B_ffn, Nn);
  k_hout<<<(unsigned)(((long)Nn*256 + 255)/256), 256, 0, stream>>>(
      hnmodh, ws + B_nt, ws + B_ffn, mask, out_h, (long)Nn * 256);

  // 10) edge residual + FF
  k_edge_res<<<eb, 256, 0, stream>>>(eain, ws + B_et, ws + B_n2e, hemodh, E);
  gemm_wmma<64,128,4,true,_Float16><<<gemm_blocks(E,128,4),256,0,stream>>>(
      ADirectH{hemodh, 64}, W_ff3, P[FF3_B], ff3h, E);
  gemm_wmma<128,64,4,false,float><<<gemm_blocks(E,64,4),256,0,stream>>>(
      ADirectH{ff3h, 128}, W_ff4, P[FF4_B], ws + B_ffe, E);
  k_eout<<<(unsigned)(((long)E*64 + 255)/256), 256, 0, stream>>>(
      hemodh, ws + B_et, ws + B_ffe, out_edge, (long)E * 64);

  // 11) equivariant coordinate update (B_e0 reused as eq_in out; B_e1 as f16 pair)
  gemm_wmma<640,256,4,false,float><<<gemm_blocks(E,256,4),256,0,stream>>>(
      AEq{out_h, out_edge, ws + B_dist, ei, ei + E}, W_eqi, P[EQI_B], ws + B_e0, E);
  k_eq_mod<<<eb, 256, 0, stream>>>(ws + B_e0, tsh, eqmh, E);
  gemm_wmma<256,256,4,true,_Float16><<<gemm_blocks(E,256,4),256,0,stream>>>(
      ADirectH{eqmh, 256}, W_eqc1, P[EQC1_B], inv2h, E);
  k_pos_init<<<(unsigned)(((long)Nn*3 + 255)/256), 256, 0, stream>>>(
      pos, out_pos, (long)Nn * 3);
  k_eq_final<<<eb, 256, 0, stream>>>(inv2h, P[EQC2_W], pos, ei,
      P[EQ_SCALE], out_pos, E);
}
